// MultiheadAttention_12876311954104
// MI455X (gfx1250) — matrix operand order repaired and checked by reading, not compiled, not run on hardware
//
#include <hip/hip_runtime.h>
#include <hip/hip_bf16.h>

typedef __bf16 bf16;
typedef __attribute__((ext_vector_type(16))) __bf16 v16bf;
typedef __attribute__((ext_vector_type(8)))  float  v8f;

// Problem dims (fixed by the reference)
#define N_B 32
#define Q_B 64
#define F_B 16
#define L_B 196
#define H_B 12
#define D_B 64
#define E_B 768
#define K_B 3136
#define LP  224   // patches padded to a multiple of the WMMA K-depth (32)

// ---------------------------------------------------------------------------
// WMMA fragment helpers (wave32, 16x16x32 bf16 -> f32)
// A (16x32, 16-bit, row-major lda): lanes 0-15 hold M=lane, K = 0..15;
// lanes 16-31 hold M=lane-16, K = 16..31   (contiguous, matches B)
// ---------------------------------------------------------------------------
__device__ __forceinline__ v16bf load_a_frag(const bf16* A, int lda) {
  const int lane = threadIdx.x & 31;
  const int hf   = lane >> 4;
  const int r    = lane & 15;
  const bf16* base = A + r * lda + hf * 16;
  v16bf a;
#pragma unroll
  for (int j = 0; j < 16; ++j) a[j] = base[j];         // K = hf*16 + 0..15
  return a;
}

// B (32x16, 16-bit) supplied "transposed" as Bt[col][k] (row-major ldb):
// lanes 0-15: N=lane, K=0..15 ; lanes 16-31: N=lane-16, K=16..31
__device__ __forceinline__ v16bf load_b_frag(const bf16* Bt, int ldb) {
  const int lane = threadIdx.x & 31;
  const int hf   = lane >> 4;
  const int r    = lane & 15;
  const bf16* base = Bt + r * ldb + hf * 16;
  v16bf b;
#pragma unroll
  for (int j = 0; j < 16; ++j) b[j] = base[j];
  return b;
}

__device__ __forceinline__ v8f wmma_bf16f32(v16bf a, v16bf b, v8f c) {
  // (neg_a, A, neg_b, B, c_mod, C, reuse_a, reuse_b)
  return __builtin_amdgcn_wmma_f32_16x16x32_bf16(false, a, false, b,
                                                 (short)0, c, false, false);
}

// Async global->LDS copy of one 16-byte chunk per lane (no VGPR data path,
// tracked by ASYNCcnt).  lds_off is the wave-relative LDS byte address
// (low 32 bits of the generic pointer, per ISA aperture translation).
__device__ __forceinline__ void async_copy_b128(uint32_t lds_off, const void* gaddr) {
  asm volatile("global_load_async_to_lds_b128 %0, %1, off"
               :: "v"(lds_off), "v"(gaddr) : "memory");
}
__device__ __forceinline__ void wait_asynccnt0() {
  asm volatile("s_wait_asynccnt 0" ::: "memory");
}

// ---------------------------------------------------------------------------
// Projection GEMM: C(M=2048, N=768) = A(M,768) * W(N,768)^T, bf16 WMMA, f32 acc
// MODE 0: out_proj -> write f32 to outF
// MODE 1: in_proj  -> write f32 qs to outF AND bf16 qs to outB in (n,h,q,d)
// ---------------------------------------------------------------------------
template<int MODE>
__global__ void __launch_bounds__(256)
proj_gemm(const float* __restrict__ A, const float* __restrict__ W,
          float* __restrict__ outF, bf16* __restrict__ outB) {
  __shared__ bf16 sA[64 * 32];
  __shared__ bf16 sB[64 * 32];
  const int tid    = threadIdx.x;
  const int w      = tid >> 5;
  const int rt     = w >> 1;          // row sub-tile 0..3
  const int ctBase = (w & 1) * 2;     // col sub-tiles {ctBase, ctBase+1}
  const int rowBlk = blockIdx.x * 64;
  const int colBlk = blockIdx.y * 64;

  v8f acc[2];
#pragma unroll
  for (int j = 0; j < 2; ++j)
#pragma unroll
    for (int i = 0; i < 8; ++i) acc[j][i] = 0.f;

  for (int k0 = 0; k0 < E_B; k0 += 32) {
    // stage 64x32 f32 slices of A and W as bf16, 128-bit global loads
    for (int i = tid; i < 64 * 32 / 4; i += 256) {
      const int rr = i >> 3, k4 = (i & 7) * 4;
      const float4 a4 = *(const float4*)&A[(size_t)(rowBlk + rr) * E_B + k0 + k4];
      const float4 w4 = *(const float4*)&W[(size_t)(colBlk + rr) * E_B + k0 + k4];
      bf16* da = sA + rr * 32 + k4;
      bf16* dw = sB + rr * 32 + k4;
      da[0] = (bf16)a4.x; da[1] = (bf16)a4.y; da[2] = (bf16)a4.z; da[3] = (bf16)a4.w;
      dw[0] = (bf16)w4.x; dw[1] = (bf16)w4.y; dw[2] = (bf16)w4.z; dw[3] = (bf16)w4.w;
    }
    __syncthreads();
    v16bf a  = load_a_frag(sA + rt * 16 * 32, 32);
    v16bf b0 = load_b_frag(sB + (ctBase + 0) * 16 * 32, 32);
    v16bf b1 = load_b_frag(sB + (ctBase + 1) * 16 * 32, 32);
    acc[0] = wmma_bf16f32(a, b0, acc[0]);
    acc[1] = wmma_bf16f32(a, b1, acc[1]);
    __syncthreads();
  }

  const int lane = tid & 31;
  const int hf   = lane >> 4;
  const int r    = lane & 15;
#pragma unroll
  for (int j = 0; j < 2; ++j) {
    const int col = colBlk + (ctBase + j) * 16 + r;
#pragma unroll
    for (int i = 0; i < 8; ++i) {
      const int row   = rowBlk + rt * 16 + hf * 8 + i;
      const float val = acc[j][i];
      outF[(size_t)row * E_B + col] = val;
      if (MODE == 1) {
        const int n = row >> 6, qr = row & 63;
        const int hh = col >> 6, dd = col & 63;
        outB[(((size_t)(n * H_B + hh) * Q_B + qr) * D_B) + dd] = (bf16)val;
      }
    }
  }
}

// ---------------------------------------------------------------------------
// Fused attention: one block per (n, h). Loop over frames; per frame:
// stage Kf / Vf^T in LDS (bf16), S = Q*Kf^T (WMMA), masked per-frame softmax
// (fp32), mix += P*Vf (WMMA regs). K/V read exactly once from HBM.
// ---------------------------------------------------------------------------
__global__ void __launch_bounds__(256)
attn_kernel(const float* __restrict__ kg, const float* __restrict__ vg,
            const unsigned char* __restrict__ mg,
            const bf16* __restrict__ qs_bf, float* __restrict__ mix) {
  extern __shared__ char smem[];
  bf16*  sQ  = (bf16*)(smem);                  // 64 x 64 bf16 (8 KB)
  bf16*  sKP = (bf16*)(smem + 8192);           // Kf 224x64 ; reused as P 64x224
  bf16*  sVT = (bf16*)(smem + 8192 + 28672);   // Vf^T 64 x 224
  float* sS  = (float*)(smem + 8192 + 2 * 28672); // scores 64 x 224 f32

  const int tid  = threadIdx.x;
  const int n    = blockIdx.x / H_B;
  const int h    = blockIdx.x % H_B;
  const int w    = tid >> 5;
  const int lane = tid & 31;
  const int hf   = lane >> 4;
  const int r    = lane & 15;

  // qs tile for this (n,h): already bf16 in workspace -> async copy to LDS
  {
    const char* src = (const char*)(qs_bf + (size_t)(n * H_B + h) * Q_B * D_B);
    const uint32_t dst0 = (uint32_t)(uintptr_t)sQ;   // LDS byte offset
#pragma unroll
    for (int i = 0; i < 2; ++i) {                    // 8192 B = 2 * 256 * 16 B
      const int byteoff = (i * 256 + tid) * 16;
      async_copy_b128(dst0 + byteoff, src + byteoff);
    }
    wait_asynccnt0();
  }

  const int mrt = w >> 1;        // mix row sub-tile
  const int mct = (w & 1) * 2;   // mix col sub-tiles {mct, mct+1}
  v8f macc[2];
#pragma unroll
  for (int j = 0; j < 2; ++j)
#pragma unroll
    for (int i = 0; i < 8; ++i) macc[j][i] = 0.f;

  const float* kf0 = kg + ((size_t)n * K_B * H_B + h) * D_B;
  const float* vf0 = vg + ((size_t)n * K_B * H_B + h) * D_B;

  for (int f = 0; f < F_B; ++f) {
    __syncthreads(); // previous iteration's readers of sKP/sVT done; sQ visible
    const size_t pbase = (size_t)f * L_B;
    // stage Kf [patch][dd] and Vf^T [dd][patch] as bf16; 128-bit global loads
    for (int i = tid; i < LP * D_B / 4; i += 256) {
      const int p  = i >> 4;          // patch (16 float4 per 64-wide row)
      const int d4 = (i & 15) * 4;
      float4 kv = make_float4(0.f, 0.f, 0.f, 0.f);
      float4 vv = make_float4(0.f, 0.f, 0.f, 0.f);
      if (p < L_B) {
        const size_t off = (pbase + p) * (size_t)(H_B * D_B) + d4;
        kv = *(const float4*)(kf0 + off);
        vv = *(const float4*)(vf0 + off);
      }
      bf16* kd = sKP + p * D_B + d4;
      kd[0] = (bf16)kv.x; kd[1] = (bf16)kv.y; kd[2] = (bf16)kv.z; kd[3] = (bf16)kv.w;
      sVT[(d4 + 0) * LP + p] = (bf16)vv.x;
      sVT[(d4 + 1) * LP + p] = (bf16)vv.y;
      sVT[(d4 + 2) * LP + p] = (bf16)vv.z;
      sVT[(d4 + 3) * LP + p] = (bf16)vv.w;
    }
    if (f + 1 < F_B) { // prefetch next frame's K/V (global_prefetch_b8)
      const size_t nxt = (pbase + L_B) * (size_t)(H_B * D_B);
      __builtin_prefetch(kf0 + nxt + (size_t)tid * 16, 0, 0);
      __builtin_prefetch(vf0 + nxt + (size_t)tid * 16, 0, 0);
    }
    __syncthreads();

    // S(64 x 224) = (qs @ Kf^T) * D^-1/2 ; 4x14 = 56 tiles over 8 waves
    for (int t = w; t < 56; t += 8) {
      const int rt = t & 3;
      const int ct = t >> 2;
      v8f acc;
#pragma unroll
      for (int i = 0; i < 8; ++i) acc[i] = 0.f;
#pragma unroll
      for (int k0 = 0; k0 < D_B; k0 += 32) {
        v16bf a = load_a_frag(sQ  + rt * 16 * D_B + k0, D_B);
        v16bf b = load_b_frag(sKP + ct * 16 * D_B + k0, D_B);
        acc = wmma_bf16f32(a, b, acc);
      }
#pragma unroll
      for (int i = 0; i < 8; ++i) {
        const int row = rt * 16 + hf * 8 + i;
        const int col = ct * 16 + r;
        sS[row * LP + col] = acc[i] * 0.125f; // 1/sqrt(64)
      }
    }
    __syncthreads();

    // masked per-frame softmax over 196 patches; 4 lanes per query row
    {
      const int row  = tid >> 2;
      const int part = tid & 3;
      const unsigned char* mrow = mg + (size_t)n * K_B + f * L_B;
      const int c0 = part * 49, c1 = c0 + 49;
      float* srow = sS + row * LP;
      float mx = -__builtin_inff();
      for (int c = c0; c < c1; ++c)
        if (mrow[c]) mx = fmaxf(mx, srow[c]);
      mx = fmaxf(mx, __shfl_xor(mx, 1));
      mx = fmaxf(mx, __shfl_xor(mx, 2));
      float sum = 0.f;
      for (int c = c0; c < c1; ++c) {
        const float p = mrow[c] ? __expf(srow[c] - mx) : 0.f;
        srow[c] = p;
        sum += p;
      }
      sum += __shfl_xor(sum, 1);
      sum += __shfl_xor(sum, 2);
      const float inv = 1.f / sum;        // >=1 valid patch guaranteed
      bf16* prow = sKP + row * LP;        // reuse Kf buffer as P (64 x 224)
      for (int c = c0; c < c1; ++c) prow[c] = (bf16)(srow[c] * inv);
    }
    for (int i = tid; i < Q_B * (LP - L_B); i += 256) { // zero padded cols
      const int row = i / (LP - L_B);
      const int c   = L_B + i % (LP - L_B);
      sKP[row * LP + c] = (bf16)0.f;
    }
    __syncthreads();

    // mix(64x64) += P(64x224) @ Vf(224x64)
#pragma unroll
    for (int k0 = 0; k0 < LP; k0 += 32) {
      v16bf a  = load_a_frag(sKP + mrt * 16 * LP + k0, LP);
      v16bf b0 = load_b_frag(sVT + (mct + 0) * 16 * LP + k0, LP);
      v16bf b1 = load_b_frag(sVT + (mct + 1) * 16 * LP + k0, LP);
      macc[0] = wmma_bf16f32(a, b0, macc[0]);
      macc[1] = wmma_bf16f32(a, b1, macc[1]);
    }
  }

  // write mix as (N*Q, E) row-major fp32 for the out_proj GEMM
#pragma unroll
  for (int j = 0; j < 2; ++j) {
    const int col = (mct + j) * 16 + r;
#pragma unroll
    for (int i = 0; i < 8; ++i) {
      const int qrow = mrt * 16 + hf * 8 + i;
      mix[((size_t)(n * Q_B + qrow)) * E_B + h * D_B + col] = macc[j][i];
    }
  }
}

// ---------------------------------------------------------------------------
extern "C" void kernel_launch(void* const* d_in, const int* in_sizes, int n_in,
                              void* d_out, int out_size, void* d_ws, size_t ws_size,
                              hipStream_t stream) {
  (void)in_sizes; (void)n_in; (void)out_size; (void)ws_size;
  const float* q   = (const float*)d_in[0];                 // (N,Q,E) f32
  const float* k   = (const float*)d_in[1];                 // (N,K,H,D) f32
  const float* v   = (const float*)d_in[2];                 // (N,K,H,D) f32
  const unsigned char* m = (const unsigned char*)d_in[3];   // (N,K) bool
  const float* W_in  = (const float*)d_in[4];               // (E,E) f32
  const float* W_out = (const float*)d_in[5];               // (E,E) f32

  float* out    = (float*)d_out;                         // (N,Q,E)
  float* qs_out = out + (size_t)N_B * Q_B * E_B;         // (N,1,Q,E)

  bf16*  ws_qs  = (bf16*)d_ws;                                        // 3.0 MB
  float* ws_mix = (float*)((char*)d_ws + (size_t)N_B * Q_B * E_B * 2); // 6.0 MB

  const dim3 gg(N_B * Q_B / 64, E_B / 64); // (32, 12)

  // 1) qs = q @ W_in^T  (fp32 copy -> d_out, bf16 copy -> workspace)
  proj_gemm<1><<<gg, 256, 0, stream>>>(q, W_in, qs_out, ws_qs);

  // 2) fused frame-masked attention, one block per (n, h)
  const size_t attn_lds = 8192 + 2 * (size_t)(LP * D_B * 2) + (size_t)(Q_B * LP * 4);
  attn_kernel<<<N_B * H_B, 256, attn_lds, stream>>>(k, v, m, ws_qs, ws_mix);

  // 3) out = mix @ W_out^T
  proj_gemm<0><<<gg, 256, 0, stream>>>(ws_mix, W_out, out, nullptr);
}
